// KDE2D_61641370632985
// MI455X (gfx1250) — matrix operand order repaired and checked by reading, not compiled, not run on hardware
//
#include <hip/hip_runtime.h>
#include <hip/hip_bf16.h>

typedef __attribute__((ext_vector_type(16))) _Float16 v16h;
typedef __attribute__((ext_vector_type(8)))  _Float16 v8h;
typedef __attribute__((ext_vector_type(8)))  float    v8f;

#define GX 128
#define GY 128
#define NPTS 512
#define CHUNK 32
#define NCHUNK (NPTS / CHUNK)
#define EX_STRIDE 40    // halves per g-row: 32 data + 8 pad (80B, 16B aligned)
#define EYT_STRIDE 136  // halves per n-row: 128 data + 8 pad (272B, 16B aligned)

#if defined(__has_builtin)
#if __has_builtin(__builtin_amdgcn_sched_group_barrier)
#define SCHED_GROUP(mask, cnt, id) __builtin_amdgcn_sched_group_barrier((mask), (cnt), (id))
#endif
#endif
#ifndef SCHED_GROUP
#define SCHED_GROUP(mask, cnt, id)
#endif
// AMDGPU sched_group_barrier masks: 0x100 = DS read, 0x008 = matrix (WMMA)

__global__ __launch_bounds__(256)
void KDE2D_61641370632985_kernel(const float* __restrict__ A,
                                 const float* __restrict__ bwp,
                                 float* __restrict__ out)
{
    __shared__ float xs[NPTS];
    __shared__ float ys[NPTS];
    __shared__ _Float16 exl[2][GX * EX_STRIDE];      // double-buffered [g][n] f16
    __shared__ _Float16 eyt[2][CHUNK * EYT_STRIDE];  // double-buffered [n][h] f16
    __shared__ float red[8][4];

    const int tid  = threadIdx.x;
    const int lane = tid & 31;
    const int wave = tid >> 5;
    const int bt   = blockIdx.x;            // 0..1023 = b*64 + t

    const float bw     = bwp[0];
    const float inv_b2 = 1.0f / (bw * bw);
    const float c2     = -0.72134752044448170f * inv_b2; // -0.5*log2(e)/bw^2
    const float norm   =  0.15915494309189535f * inv_b2; // 1/(2*pi*bw^2)
    const float gstep  = 10.0f / 127.0f;

    // ---------- Phase 0: load points, mean/std (ddof=1), standardize ----------
    const float* Ab = A + (size_t)bt * (NPTS * 2);
    float4 p = ((const float4*)Ab)[tid];    // points 2*tid, 2*tid+1: (x0,y0,x1,y1)
    float sx  = p.x + p.z,          sy  = p.y + p.w;
    float sxx = p.x*p.x + p.z*p.z,  syy = p.y*p.y + p.w*p.w;
    #pragma unroll
    for (int off = 16; off > 0; off >>= 1) {
        sx  += __shfl_xor(sx,  off, 32);
        sy  += __shfl_xor(sy,  off, 32);
        sxx += __shfl_xor(sxx, off, 32);
        syy += __shfl_xor(syy, off, 32);
    }
    if (lane == 0) { red[wave][0]=sx; red[wave][1]=sy; red[wave][2]=sxx; red[wave][3]=syy; }
    __syncthreads();
    sx = sy = sxx = syy = 0.0f;
    #pragma unroll
    for (int wv = 0; wv < 8; ++wv) {
        sx += red[wv][0]; sy += red[wv][1]; sxx += red[wv][2]; syy += red[wv][3];
    }
    const float mx = sx * (1.0f / NPTS), my = sy * (1.0f / NPTS);
    const float vx = (sxx - NPTS * mx * mx) * (1.0f / (NPTS - 1));
    const float vy = (syy - NPTS * my * my) * (1.0f / (NPTS - 1));
    const float isx = rsqrtf(vx), isy = rsqrtf(vy);
    xs[2*tid]   = (p.x - mx) * isx;
    xs[2*tid+1] = (p.z - mx) * isx;
    ys[2*tid]   = (p.y - my) * isy;
    ys[2*tid+1] = (p.w - my) * isy;
    __syncthreads();

    // ---- chunk builder: 32 exps/thread -> f16 chunks in LDS buffer `buf` ----
    auto build_chunk = [&](int ck, int buf) {
        const int n0 = ck * CHUNK;
        {   // ex chunk [g 0..127][n 0..31]
            const int g  = tid & 127;
            const int nh = tid >> 7;                 // n-range 0..15 / 16..31
            const float xg = -5.0f + g * gstep;
            const float4* xv = (const float4*)(xs + n0 + nh * 16);
            _Float16 e[16];
            #pragma unroll
            for (int q = 0; q < 4; ++q) {
                float4 v = xv[q];
                float d;
                d = xg - v.x; e[4*q+0] = (_Float16)__builtin_amdgcn_exp2f(c2 * d * d);
                d = xg - v.y; e[4*q+1] = (_Float16)__builtin_amdgcn_exp2f(c2 * d * d);
                d = xg - v.z; e[4*q+2] = (_Float16)__builtin_amdgcn_exp2f(c2 * d * d);
                d = xg - v.w; e[4*q+3] = (_Float16)__builtin_amdgcn_exp2f(c2 * d * d);
            }
            v8h lo, hi;
            #pragma unroll
            for (int i = 0; i < 8; ++i) { lo[i] = e[i]; hi[i] = e[i + 8]; }
            _Float16* dst = exl[buf] + g * EX_STRIDE + nh * 16;
            *(v8h*)(dst)     = lo;                   // ds_store_b128
            *(v8h*)(dst + 8) = hi;
        }
        {   // ey chunk transposed [n 0..31][h 0..127]
            const int n  = tid >> 3;                 // 0..31
            const int hs = tid & 7;                  // 16-wide h segment
            const float yv = ys[n0 + n];
            _Float16 e[16];
            #pragma unroll
            for (int j = 0; j < 16; ++j) {
                const float ygv = -5.0f + (hs * 16 + j) * gstep;
                const float d = ygv - yv;
                e[j] = (_Float16)__builtin_amdgcn_exp2f(c2 * d * d);
            }
            v8h lo, hi;
            #pragma unroll
            for (int i = 0; i < 8; ++i) { lo[i] = e[i]; hi[i] = e[i + 8]; }
            _Float16* dst = eyt[buf] + n * EYT_STRIDE + hs * 16;
            *(v8h*)(dst)     = lo;
            *(v8h*)(dst + 8) = hi;
        }
    };

    const v8f vzero = {0.f,0.f,0.f,0.f,0.f,0.f,0.f,0.f};
    v8f acc[8];
    #pragma unroll
    for (int t = 0; t < 8; ++t) acc[t] = vzero;

    build_chunk(0, 0);
    __syncthreads();

    for (int ck = 0; ck < NCHUNK; ++ck) {
        const int buf = ck & 1;
        const _Float16* exb = exl[buf];
        const _Float16* eyb = eyt[buf];

        // A fragment (16-bit A 16x32 layout): lane<16 -> K{0..7,16..23}, else K{8..15,24..31}
        const int gA   = 16 * wave + (lane & 15);
        const int koff = (lane >> 4) ? 8 : 0;
        const _Float16* arow = exb + gA * EX_STRIDE + koff;
        v8h alo = *(const v8h*)(arow);
        v8h ahi = *(const v8h*)(arow + 16);
        v16h afrag = __builtin_shufflevector(alo, ahi,
            0,1,2,3,4,5,6,7,8,9,10,11,12,13,14,15);

        // Preload ALL 8 B fragments into distinct regs so ds loads pipeline
        v16h bfrag[8];
        #pragma unroll
        for (int t = 0; t < 8; ++t) {
            const _Float16* brow = eyb + koff * EYT_STRIDE + t * 16 + (lane & 15);
            #pragma unroll
            for (int s = 0; s < 16; ++s)
                bfrag[t][s] = brow[((s < 8) ? s : (s + 8)) * EYT_STRIDE];
        }
        // Back-to-back WMMAs (matrix pipe)
        #pragma unroll
        for (int t = 0; t < 8; ++t) {
            acc[t] = __builtin_amdgcn_wmma_f32_16x16x32_f16(
                false, afrag, false, bfrag[t],
                (short)0, acc[t], false, false);
        }
        // Pin the pipeline: group 18 of the 130 fragment DS reads, then the
        // 8 WMMAs. Scheduler can then use staged s_wait_dscnt instead of 8x "0".
        SCHED_GROUP(0x100, 18, 0);   // 18 DS reads; A frag is 2 ds_load_b128, B frags are 128 16-bit loads
        SCHED_GROUP(0x008, 8, 0);    // 8 v_wmma back-to-back

        // Overlap: generate next chunk's exps (trans pipe) while WMMAs drain
        if (ck + 1 < NCHUNK) build_chunk(ck + 1, buf ^ 1);
        __syncthreads();
    }

    // ---------- store: C layout VGPR r -> M = r + 8*(lane>=16), N = lane&15 ----------
    float* outp = out + (size_t)bt * (GX * GY);
    const int mrow = (lane >> 4) * 8;
    const int ncol = lane & 15;
    #pragma unroll
    for (int t = 0; t < 8; ++t) {
        #pragma unroll
        for (int r = 0; r < 8; ++r) {
            const int g = 16 * wave + mrow + r;
            const int h = 16 * t + ncol;
            outp[g * GY + h] = acc[t][r] * norm;
        }
    }
}

extern "C" void kernel_launch(void* const* d_in, const int* in_sizes, int n_in,
                              void* d_out, int out_size, void* d_ws, size_t ws_size,
                              hipStream_t stream) {
    (void)in_sizes; (void)n_in; (void)out_size; (void)d_ws; (void)ws_size;
    const float* A  = (const float*)d_in[0];   // [16,64,512,2] f32
    const float* bw = (const float*)d_in[1];   // scalar f32
    float* out      = (float*)d_out;           // [16,64,128,128] f32
    dim3 grid(16 * 64);                        // one workgroup per (b,t)
    dim3 block(256);                           // 8 wave32
    KDE2D_61641370632985_kernel<<<grid, block, 0, stream>>>(A, bw, out);
}
